// GNNEncoder_75488345195247
// MI455X (gfx1250) — matrix operand order repaired and checked by reading, not compiled, not run on hardware
//
#include <hip/hip_runtime.h>
#include <hip/hip_bf16.h>
#include <stdint.h>

#define NODES  50000
#define EDGES  1600000
#define GRAPHS 128
#define CH     64       // H == CIN == 64
#define BN_EPS 1e-5f

typedef __attribute__((ext_vector_type(16))) __bf16       v16bf;
typedef __attribute__((ext_vector_type(8)))  float        v8f;
typedef __attribute__((ext_vector_type(8)))  unsigned int v8u;

// Native bf16 converts: compiler selects HW packed cvt (v_cvt_*bf16_f32)
__device__ __forceinline__ unsigned short bf16s(float f) {
  __bf16 h = (__bf16)f;
  return __builtin_bit_cast(unsigned short, h);
}
__device__ __forceinline__ unsigned int packbf(float a, float b) {
  return (unsigned int)bf16s(a) | ((unsigned int)bf16s(b) << 16);
}
__device__ __forceinline__ v8f wmma_bf16(v8u a, v8u b, v8f c) {
  return __builtin_amdgcn_wmma_f32_16x16x32_bf16(
      false, __builtin_bit_cast(v16bf, a),
      false, __builtin_bit_cast(v16bf, b),
      (short)0, c, false, false);
}
// wave-local LDS producer->consumer fence (waves run independent tile streams,
// so no workgroup barrier; CDNA5 split DS counter wait instead)
__device__ __forceinline__ void wave_lds_sync() {
  __builtin_amdgcn_wave_barrier();
  asm volatile("s_wait_dscnt 0" ::: "memory");
  __builtin_amdgcn_wave_barrier();
}
__device__ __forceinline__ void atomicMaxF(float* addr, float v) {
  if (v >= 0.0f) atomicMax((int*)addr, __float_as_int(v));
  else           atomicMin((unsigned int*)addr, (unsigned int)__float_as_int(v));
}
__device__ __forceinline__ void ld8(const float* __restrict__ p, float* r) {
  float4 a = *(const float4*)p;
  float4 b = *(const float4*)(p + 4);
  r[0] = a.x; r[1] = a.y; r[2] = a.z; r[3] = a.w;
  r[4] = b.x; r[5] = b.y; r[6] = b.z; r[7] = b.w;
}

// ---------------------------------------------------------------------------
// Fused EdgeConv layer: gather -> (16x128)@(128x64) -> relu -> (16x64)@(64x64)
// -> relu -> scatter atomic-max.  One wave per 16-edge tile, bf16 WMMA.
// ---------------------------------------------------------------------------
__global__ __launch_bounds__(256) void edgeconv_wmma(
    const float* __restrict__ xin,                       // [NODES, 64]
    const int* __restrict__ src, const int* __restrict__ dst,
    const float* __restrict__ Wa, const float* __restrict__ ba,   // [128,64],[64]
    const float* __restrict__ Wb, const float* __restrict__ bb,   // [64,64],[64]
    float* __restrict__ out)                             // [NODES,64] zeroed
{
  // fragment-ready weight tiles: [frag][lane][v] packed bf16 pairs
  __shared__ __align__(16) unsigned int   wa_frag[16 * 32 * 8];   // 16 KB
  __shared__ __align__(16) unsigned int   wb_frag[8 * 32 * 8];    //  8 KB
  __shared__ __align__(16) unsigned short hbuf[8 * 16 * 64];      // 16 KB

  const int tid = threadIdx.x;

  // Stage Wa into B-fragment layout: frag=(k,n), lane=(half,nn), K=k*32+(v>>2)*16+half*8+(v&3)*2+s
  for (int i = tid; i < 16 * 32 * 8; i += 256) {
    int v = i & 7, lane = (i >> 3) & 31, frag = i >> 8;
    int k = frag >> 2, n = frag & 3;
    int half = lane >> 4, nn = n * 16 + (lane & 15);
    int K0 = k * 32 + (v >> 2) * 16 + half * 8 + (v & 3) * 2;
    wa_frag[i] = packbf(Wa[K0 * 64 + nn], Wa[(K0 + 1) * 64 + nn]);
  }
  for (int i = tid; i < 8 * 32 * 8; i += 256) {
    int v = i & 7, lane = (i >> 3) & 31, frag = i >> 8;
    int k = frag >> 2, n = frag & 3;
    int half = lane >> 4, nn = n * 16 + (lane & 15);
    int K0 = k * 32 + (v >> 2) * 16 + half * 8 + (v & 3) * 2;
    wb_frag[i] = packbf(Wb[K0 * 64 + nn], Wb[(K0 + 1) * 64 + nn]);
  }
  __syncthreads();

  const int lane = tid & 31;
  const int wave = tid >> 5;
  const int m    = lane & 15;
  const int half = lane >> 4;
  unsigned short* hrow = &hbuf[wave * (16 * 64)];

  const int ntiles = (EDGES + 15) >> 4;
  const int tstep  = gridDim.x * 8;
  for (int t = blockIdx.x * 8 + wave; t < ntiles; t += tstep) {
    const int e  = t * 16 + m;
    const int ec = e < EDGES ? e : EDGES - 1;
    const int di = dst[ec];
    const int si = src[ec];
    const float* xd = xin + (size_t)di * 64;
    const float* xs = xin + (size_t)si * 64;

    // prefetch next tile's edge indices (global_prefetch_b8)
    {
      const int tn = t + tstep;
      if (tn < ntiles) {
        __builtin_prefetch(dst + tn * 16 + m, 0, 0);
        __builtin_prefetch(src + tn * 16 + m, 0, 0);
      }
    }

    // ---- A = [x_i | x_j - x_i] as 4 bf16 16x32 fragments ----
    v8u afr[4];
#pragma unroll
    for (int k = 0; k < 4; ++k) {
      const int col = (k & 1) * 32 + half * 8;   // k0,k1: x_i cols; k2,k3: x_j-x_i cols
      float lo[8], hi[8];
      ld8(xd + col, lo);
      ld8(xd + col + 16, hi);
      if (k >= 2) {
        float l2[8], h2[8];
        ld8(xs + col, l2);
        ld8(xs + col + 16, h2);
#pragma unroll
        for (int j = 0; j < 8; ++j) { lo[j] = l2[j] - lo[j]; hi[j] = h2[j] - hi[j]; }
      }
      v8u a;
#pragma unroll
      for (int v = 0; v < 4; ++v) {
        a[v]     = packbf(lo[2 * v], lo[2 * v + 1]);
        a[v + 4] = packbf(hi[2 * v], hi[2 * v + 1]);
      }
      afr[k] = a;
    }

    // ---- GEMM1: (16x128)@(128x64), +bias, ReLU -> LDS bf16 tile ----
#pragma unroll
    for (int n = 0; n < 4; ++n) {
      v8f c = {};
#pragma unroll
      for (int k = 0; k < 4; ++k) {
        const unsigned int* bp = &wa_frag[((k * 4 + n) * 32 + lane) * 8];
        v8u b;
#pragma unroll
        for (int v = 0; v < 8; ++v) b[v] = bp[v];
        c = wmma_bf16(afr[k], b, c);
      }
      const float bias = ba[n * 16 + m];
#pragma unroll
      for (int v = 0; v < 8; ++v) {
        float h = c[v] + bias;
        h = h > 0.0f ? h : 0.0f;
        hrow[(v + 8 * half) * 64 + n * 16 + m] = bf16s(h);
      }
    }
    wave_lds_sync();

    // ---- Rebuild A2 (16x64 hidden) in A-fragment layout via b128 LDS reads ----
    v8u a2[2];
#pragma unroll
    for (int k = 0; k < 2; ++k) {
      const int off = m * 64 + k * 32 + half * 8;
      uint4 q0 = *(const uint4*)&hrow[off];
      uint4 q1 = *(const uint4*)&hrow[off + 16];
      v8u a;
      a[0] = q0.x; a[1] = q0.y; a[2] = q0.z; a[3] = q0.w;
      a[4] = q1.x; a[5] = q1.y; a[6] = q1.z; a[7] = q1.w;
      a2[k] = a;
    }

    // dst rows this lane scatters (D-layout rows M = v + 8*half)
    int drow[8];
#pragma unroll
    for (int v = 0; v < 8; ++v) {
      int e2 = t * 16 + v + 8 * half;
      drow[v] = dst[e2 < EDGES ? e2 : EDGES - 1];
    }

    // ---- GEMM2: (16x64)@(64x64), +bias, ReLU -> scatter atomic max ----
#pragma unroll
    for (int n = 0; n < 4; ++n) {
      v8f c = {};
#pragma unroll
      for (int k = 0; k < 2; ++k) {
        const unsigned int* bp = &wb_frag[((k * 4 + n) * 32 + lane) * 8];
        v8u b;
#pragma unroll
        for (int v = 0; v < 8; ++v) b[v] = bp[v];
        c = wmma_bf16(a2[k], b, c);
      }
      const float bias = bb[n * 16 + m];
#pragma unroll
      for (int v = 0; v < 8; ++v) {
        float h = c[v] + bias;
        h = h > 0.0f ? h : 0.0f;
        const int e2 = t * 16 + v + 8 * half;
        if (e2 < EDGES) {
          // post-ReLU values are >= 0: int-punned bits are monotonic
          atomicMax((int*)&out[(size_t)drow[v] * 64 + n * 16 + m],
                    __float_as_int(h));
        }
      }
    }
  }
}

// ---------------------------------------------------------------------------
// BatchNorm: per-channel sum / sumsq reduction, then normalize
// ---------------------------------------------------------------------------
__global__ __launch_bounds__(256) void bn_stats_kernel(
    const float* __restrict__ x, float* __restrict__ stats /* [128] zeroed */) {
  __shared__ float ssum[64], ssq[64];
  const int tid = threadIdx.x;
  if (tid < 64) { ssum[tid] = 0.0f; ssq[tid] = 0.0f; }
  __syncthreads();
  const long total = (long)NODES * CH;
  for (long i = (long)blockIdx.x * blockDim.x + tid; i < total;
       i += (long)gridDim.x * blockDim.x) {
    float v = x[i];
    int c = (int)(i & 63);
    atomicAdd(&ssum[c], v);
    atomicAdd(&ssq[c], v * v);
  }
  __syncthreads();
  if (tid < 64) {
    atomicAdd(&stats[tid],      ssum[tid]);
    atomicAdd(&stats[64 + tid], ssq[tid]);
  }
}

__global__ __launch_bounds__(256) void bn_apply_kernel(
    const float* __restrict__ x, const float* __restrict__ stats,
    const float* __restrict__ gamma, const float* __restrict__ beta,
    float* __restrict__ y) {
  const long total = (long)NODES * CH;
  long i = (long)blockIdx.x * blockDim.x + threadIdx.x;
  if (i >= total) return;
  int c = (int)(i & 63);
  float mu  = stats[c] * (1.0f / NODES);
  float var = stats[64 + c] * (1.0f / NODES) - mu * mu;
  y[i] = (x[i] - mu) * rsqrtf(var + BN_EPS) * gamma[c] + beta[c];
}

// ---------------------------------------------------------------------------
// Graph pooling: mean + max over batch segments
// ---------------------------------------------------------------------------
__global__ __launch_bounds__(256) void pool_init_kernel(
    float* __restrict__ sums, float* __restrict__ mx, float* __restrict__ cnt) {
  int i = blockIdx.x * blockDim.x + threadIdx.x;
  if (i < GRAPHS * CH) { sums[i] = 0.0f; mx[i] = -INFINITY; }
  if (i < GRAPHS) cnt[i] = 0.0f;
}

__global__ __launch_bounds__(256) void pool_accum_kernel(
    const float* __restrict__ h, const int* __restrict__ batch,
    float* __restrict__ sums, float* __restrict__ mx, float* __restrict__ cnt) {
  const long total = (long)NODES * CH;
  for (long i = (long)blockIdx.x * blockDim.x + threadIdx.x; i < total;
       i += (long)gridDim.x * blockDim.x) {
    int node = (int)(i >> 6);
    int c    = (int)(i & 63);
    int g    = batch[node];
    float v  = h[i];
    atomicAdd(&sums[g * CH + c], v);
    atomicMaxF(&mx[g * CH + c], v);
    if (c == 0) atomicAdd(&cnt[g], 1.0f);
  }
}

__global__ __launch_bounds__(256) void pool_final_kernel(
    const float* __restrict__ sums, const float* __restrict__ mx,
    const float* __restrict__ cnt, float* __restrict__ xg) {
  int i = blockIdx.x * blockDim.x + threadIdx.x;
  if (i >= GRAPHS * CH) return;
  int g = i >> 6, c = i & 63;
  xg[g * (2 * CH) + c]      = sums[i] / fmaxf(cnt[g], 1.0f);
  xg[g * (2 * CH) + CH + c] = mx[i];
}

// ---------------------------------------------------------------------------
extern "C" void kernel_launch(void* const* d_in, const int* in_sizes, int n_in,
                              void* d_out, int out_size, void* d_ws, size_t ws_size,
                              hipStream_t stream) {
  const float* x   = (const float*)d_in[0];
  const int* eidx  = (const int*)d_in[1];
  const int* batch = (const int*)d_in[2];
  const float* W1a = (const float*)d_in[3];
  const float* b1a = (const float*)d_in[4];
  const float* W1b = (const float*)d_in[5];
  const float* b1b = (const float*)d_in[6];
  const float* g1  = (const float*)d_in[7];
  const float* be1 = (const float*)d_in[8];
  const float* W2a = (const float*)d_in[9];
  const float* b2a = (const float*)d_in[10];
  const float* W2b = (const float*)d_in[11];
  const float* b2b = (const float*)d_in[12];
  const float* g2  = (const float*)d_in[13];
  const float* be2 = (const float*)d_in[14];
  const int* src = eidx;
  const int* dst = eidx + EDGES;

  float* hout = (float*)d_out;                    // [NODES*64]
  float* xg   = hout + (size_t)NODES * CH;        // [GRAPHS*128]

  float* A     = (float*)d_ws;                    // conv scratch (scatter-max dest)
  float* B     = A + (size_t)NODES * CH;          // BN1 output
  float* stats = B + (size_t)NODES * CH;          // 128 floats
  float* psum  = stats + 128;                     // G*64
  float* pmax  = psum + GRAPHS * CH;              // G*64
  float* pcnt  = pmax + GRAPHS * CH;              // G

  const dim3 blk(256);
  const int elemBlocks = (NODES * CH + 255) / 256;

  // ----- layer 1 -----
  hipMemsetAsync(A, 0, (size_t)NODES * CH * sizeof(float), stream);
  edgeconv_wmma<<<1024, blk, 0, stream>>>(x, src, dst, W1a, b1a, W1b, b1b, A);
  hipMemsetAsync(stats, 0, 128 * sizeof(float), stream);
  bn_stats_kernel<<<512, blk, 0, stream>>>(A, stats);
  bn_apply_kernel<<<elemBlocks, blk, 0, stream>>>(A, stats, g1, be1, B);

  // ----- layer 2 -----
  hipMemsetAsync(A, 0, (size_t)NODES * CH * sizeof(float), stream);
  edgeconv_wmma<<<1024, blk, 0, stream>>>(B, src, dst, W2a, b2a, W2b, b2b, A);
  hipMemsetAsync(stats, 0, 128 * sizeof(float), stream);
  bn_stats_kernel<<<512, blk, 0, stream>>>(A, stats);
  bn_apply_kernel<<<elemBlocks, blk, 0, stream>>>(A, stats, g2, be2, hout);

  // ----- pooling -----
  pool_init_kernel<<<(GRAPHS * CH + 255) / 256, blk, 0, stream>>>(psum, pmax, pcnt);
  pool_accum_kernel<<<512, blk, 0, stream>>>(hout, batch, psum, pmax, pcnt);
  pool_final_kernel<<<(GRAPHS * CH + 255) / 256, blk, 0, stream>>>(psum, pmax, pcnt, xg);

  (void)in_sizes; (void)n_in; (void)out_size; (void)ws_size;
}
